// Attention_5428838662814
// MI455X (gfx1250) — matrix operand order repaired and checked by reading, not compiled, not run on hardware
//
#include <hip/hip_runtime.h>

typedef __attribute__((ext_vector_type(16))) _Float16 v16h;
typedef __attribute__((ext_vector_type(8)))  _Float16 v8h;
typedef __attribute__((ext_vector_type(4)))  _Float16 v4h;
typedef __attribute__((ext_vector_type(8)))  float    v8f;

#define S_SRC 2048
#define S_TGT 2048
#define BATCH 16
#define HDIM  256
#define QT    128   // query rows per block (16 per wave, 8 waves)
#define KT    64    // kv rows per LDS tile / loop iteration

static __device__ __forceinline__ v16h cat8(v8h lo, v8h hi) {
  return __builtin_shufflevector(lo, hi, 0,1,2,3,4,5,6,7,8,9,10,11,12,13,14,15);
}

static __device__ __forceinline__ v8f wmma_f16(v16h a, v16h b, v8f c) {
  return __builtin_amdgcn_wmma_f32_16x16x32_f16(false, a, false, b, (short)0, c,
                                                false, false);
}

// B-fragment from row-major LDS tile.
static __device__ __forceinline__ v16h load_kfrag(const _Float16* row, int koff) {
  return cat8(*(const v8h*)(row + koff), *(const v8h*)(row + koff + 8));
}

// B-fragment from XOR-swizzled V^T tile: element kv of row h stored at
// kv ^ ((h&7)<<3).  kstart is a multiple of 16.
static __device__ __forceinline__ v16h load_vfrag(const _Float16* row, int kstart,
                                                  int sw48, int sw8) {
  int blk = (kstart ^ sw48);
  v8h lo = *(const v8h*)(row + blk + sw8);
  v8h hi = *(const v8h*)(row + blk + (sw8 ^ 8));
  return cat8(lo, hi);
}

__global__ __launch_bounds__(256)
void attn_fa_kernel(const float* __restrict__ out_e,
                    const float* __restrict__ out_d,
                    float* __restrict__ outp)
{
  // 32 KB K tile (row-major [kv][h]); P staging reuses the first 16 KB.
  __shared__ __align__(16) _Float16 lds_k[KT][HDIM];
  // 32 KB XOR-swizzled V^T tile ([h][kv ^ ((h&7)<<3)]).
  __shared__ __align__(16) _Float16 lds_vt[HDIM][KT];

  const int tid  = threadIdx.x;
  const int lane = tid & 31;
  const int wave = tid >> 5;
  const int b     = blockIdx.x & (BATCH - 1);
  const int qtile = blockIdx.x >> 4;
  const int q0    = qtile * QT + wave * 16;   // first query row of this wave

  const int mA  = lane & 15;         // A-frag row (M) / B-frag column (N)
  const int kA  = (lane >> 4) * 8;   // A-frag K sub-offset (0 or 8)
  const int kB  = (lane >> 4) * 16;  // B-frag K sub-offset (0 or 16)
  const int swf = (lane & 7) << 3;   // V^T swizzle for this lane's h rows
  const int sw48 = swf & 48, sw8 = swf & 8;

  // ---------- Q fragments: 16 rows x 256, f32 -> f16, A-matrix layout ----------
  v16h qf[8];
  {
    const float* qb = out_d + (size_t)(q0 + mA) * (BATCH * HDIM) + b * HDIM;
#pragma unroll
    for (int c = 0; c < 8; ++c) {
      const float* p0 = qb + c * 32 + kB;   // K = c*32 + kB .. +7
      const float* p1 = p0 + 8;             // K = c*32 + kB + 8 .. +15
      float4 a0 = *(const float4*)(p0);
      float4 a1 = *(const float4*)(p0 + 4);
      float4 b0 = *(const float4*)(p1);
      float4 b1 = *(const float4*)(p1 + 4);
      v16h f;
      f[0]=(_Float16)a0.x;  f[1]=(_Float16)a0.y;  f[2]=(_Float16)a0.z;  f[3]=(_Float16)a0.w;
      f[4]=(_Float16)a1.x;  f[5]=(_Float16)a1.y;  f[6]=(_Float16)a1.z;  f[7]=(_Float16)a1.w;
      f[8]=(_Float16)b0.x;  f[9]=(_Float16)b0.y;  f[10]=(_Float16)b0.z; f[11]=(_Float16)b0.w;
      f[12]=(_Float16)b1.x; f[13]=(_Float16)b1.y; f[14]=(_Float16)b1.z; f[15]=(_Float16)b1.w;
      qf[c] = f;
    }
  }

  const v8f vzero = {};
  v8f o[16];                       // context accumulators: 16 h-tiles of 16
#pragma unroll
  for (int i = 0; i < 16; ++i) o[i] = vzero;
  float mrow[8], lsum[8];          // online-softmax state (row = r + 8*(lane>>4))
#pragma unroll
  for (int r = 0; r < 8; ++r) { mrow[r] = -3.0e38f; lsum[r] = 0.0f; }

  _Float16* const pstage = &lds_k[0][0] + wave * (16 * KT);

  for (int s0 = 0; s0 < S_SRC; s0 += KT) {
    __syncthreads();   // previous iteration's LDS consumers are done

    // ---- phase A: global -> lds_k row-major; enc = dir0 + dir1, f32->f16 ----
#pragma unroll
    for (int i = 0; i < 16; ++i) {
      int e   = tid + 256 * i;       // 0..4095
      int row = e >> 6;              // kv row 0..63
      int ch  = (e & 63) * 4;        // h offset, float4 granularity
      const float* src = out_e + (size_t)(s0 + row) * (BATCH * 2 * HDIM)
                       + b * (2 * HDIM) + ch;
      float4 d0 = *(const float4*)(src);
      float4 d1 = *(const float4*)(src + HDIM);
      v4h pk;
      pk[0] = (_Float16)(d0.x + d1.x);
      pk[1] = (_Float16)(d0.y + d1.y);
      pk[2] = (_Float16)(d0.z + d1.z);
      pk[3] = (_Float16)(d0.w + d1.w);
      *(v4h*)&lds_k[row][ch] = pk;
    }
    __syncthreads();

    // ---- phase B: lds_k -> lds_vt (transpose, XOR-swizzled, conflict-light) ----
    {
      const int h  = tid;                 // each thread owns one h row
      const int sw = (h & 7) << 3;
      _Float16* dst = &lds_vt[0][0] + h * KT;
#pragma unroll
      for (int j = 0; j < 16; ++j) {
        v4h q;
        q[0] = lds_k[4 * j + 0][h];
        q[1] = lds_k[4 * j + 1][h];
        q[2] = lds_k[4 * j + 2][h];
        q[3] = lds_k[4 * j + 3][h];
        *(v4h*)(dst + ((4 * j) ^ sw)) = q;   // b64 store, 2-way conflict worst case
      }
    }

    // prefetch next tile into cache while we compute on this one
    if (s0 + KT < S_SRC) {
      int row = tid >> 2;
      int ch  = (tid & 3) * 128;
      __builtin_prefetch(out_e + (size_t)(s0 + KT + row) * (BATCH * 2 * HDIM)
                         + b * (2 * HDIM) + ch, 0, 0);
    }
    __syncthreads();

    // ---- scores: S = Q * K^T ; c-outer so 4 accumulator chains interleave ----
    v8f sc[4];
#pragma unroll
    for (int n = 0; n < 4; ++n) sc[n] = vzero;

    v16h bk[4];
#pragma unroll
    for (int n = 0; n < 4; ++n)
      bk[n] = load_kfrag(&lds_k[n * 16 + mA][0], kB);

#pragma unroll
    for (int c = 0; c < 8; ++c) {
      v16h nb[4];
      if (c < 7) {
#pragma unroll
        for (int n = 0; n < 4; ++n)
          nb[n] = load_kfrag(&lds_k[n * 16 + mA][0], (c + 1) * 32 + kB);
      }
#pragma unroll
      for (int n = 0; n < 4; ++n)
        sc[n] = wmma_f16(qf[c], bk[n], sc[n]);
      if (c < 7) {
#pragma unroll
        for (int n = 0; n < 4; ++n) bk[n] = nb[n];
      }
    }

    // ---- online softmax over kv; rows live across 16 lanes of a half-wave ----
    float mnew[8];
#pragma unroll
    for (int r = 0; r < 8; ++r) {
      float mx = fmaxf(fmaxf(sc[0][r], sc[1][r]), fmaxf(sc[2][r], sc[3][r]));
      mx = fmaxf(mx, __shfl_xor(mx, 8, 16));
      mx = fmaxf(mx, __shfl_xor(mx, 4, 16));
      mx = fmaxf(mx, __shfl_xor(mx, 2, 16));
      mx = fmaxf(mx, __shfl_xor(mx, 1, 16));
      mnew[r] = fmaxf(mrow[r], mx);
    }
#pragma unroll
    for (int r = 0; r < 8; ++r) {
      float scale = __expf(mrow[r] - mnew[r]);
      mrow[r] = mnew[r];
      float rs = 0.0f;
#pragma unroll
      for (int n = 0; n < 4; ++n) {
        float p = __expf(sc[n][r] - mnew[r]);
        sc[n][r] = p;
        rs += p;
      }
      rs += __shfl_xor(rs, 8, 16);
      rs += __shfl_xor(rs, 4, 16);
      rs += __shfl_xor(rs, 2, 16);
      rs += __shfl_xor(rs, 1, 16);
      lsum[r] = lsum[r] * scale + rs;
#pragma unroll
      for (int t = 0; t < 16; ++t) o[t][r] = o[t][r] * scale;
    }

    __syncthreads();   // all waves finished reading lds_k; safe to overwrite with P

    // ---- stage P (f16) in per-wave LDS slice, row-major [16][KT] ----
#pragma unroll
    for (int n = 0; n < 4; ++n)
#pragma unroll
      for (int r = 0; r < 8; ++r)
        pstage[(r + 8 * (lane >> 4)) * KT + n * 16 + mA] = (_Float16)sc[n][r];

    asm volatile("s_wait_dscnt 0x0" ::: "memory");

    // ---- read P back as A fragments (K = 0..31, 32..63) ----
    v16h pf[2];
#pragma unroll
    for (int cc = 0; cc < 2; ++cc) {
      const _Float16* pp = pstage + mA * KT + cc * 32 + kB;
      pf[cc] = cat8(*(const v8h*)pp, *(const v8h*)(pp + 8));
    }

    // ---- context: O += P * V ; double-buffered swizzled B-fragments ----
    {
      const _Float16* vt0 = &lds_vt[0][0] + (size_t)mA * KT;
      v16h vb0 = load_vfrag(vt0, kB,      sw48, sw8);
      v16h vb1 = load_vfrag(vt0, 32 + kB, sw48, sw8);
#pragma unroll
      for (int ht = 0; ht < 16; ++ht) {
        v16h n0, n1;
        if (ht < 15) {
          const _Float16* vtn = &lds_vt[0][0] + (size_t)((ht + 1) * 16 + mA) * KT;
          n0 = load_vfrag(vtn, kB,      sw48, sw8);
          n1 = load_vfrag(vtn, 32 + kB, sw48, sw8);
        }
        o[ht] = wmma_f16(pf[0], vb0, o[ht]);
        o[ht] = wmma_f16(pf[1], vb1, o[ht]);
        if (ht < 15) { vb0 = n0; vb1 = n1; }
      }
    }
  }

  // ---- normalize and store: out[t][b][h], D-layout rows M = r + 8*(lane>>4) ----
  const int mb = (lane >> 4) * 8;
#pragma unroll
  for (int r = 0; r < 8; ++r) {
    float inv = 1.0f / lsum[r];
    float* op = outp + (size_t)(q0 + mb + r) * (BATCH * HDIM) + b * HDIM;
#pragma unroll
    for (int ht = 0; ht < 16; ++ht)
      op[ht * 16 + mA] = o[ht][r] * inv;
  }
}

extern "C" void kernel_launch(void* const* d_in, const int* in_sizes, int n_in,
                              void* d_out, int out_size, void* d_ws, size_t ws_size,
                              hipStream_t stream) {
  const float* out_e = (const float*)d_in[0];   // [2048, 16, 512] f32
  const float* out_d = (const float*)d_in[1];   // [2048, 16, 256] f32
  float* outp = (float*)d_out;                  // [2048, 16, 256] f32
  dim3 grid(BATCH * (S_TGT / QT));              // 16 batches * 16 q-tiles = 256
  attn_fa_kernel<<<grid, 256, 0, stream>>>(out_e, out_d, outp);
}
